// RNNPredictor_37589553775044
// MI455X (gfx1250) — compile-verified, matrix operand order checked by reading, not run on hardware
//
#include <hip/hip_runtime.h>
#include <hip/hip_bf16.h>
#include <cstdint>
#include <cstddef>

typedef __attribute__((ext_vector_type(16))) _Float16 f16x16;
typedef __attribute__((ext_vector_type(8)))  _Float16 f16x8;
typedef __attribute__((ext_vector_type(8)))  float    f32x8;

#define UNITS  1024
#define GATES  3072   // 3*UNITS
#define BATCH  128
#define TSTEPS 256
#define INDIM  128

// ---------------- fp32 -> f16 elementwise ----------------
__global__ void k_f32_to_f16(const float* __restrict__ src, _Float16* __restrict__ dst, int n) {
  int i = blockIdx.x * blockDim.x + threadIdx.x;
  if (i < n) dst[i] = (_Float16)src[i];
}

// ---------------- zero hidden state ----------------
__global__ void k_zero_h(float* __restrict__ h32, _Float16* __restrict__ h16) {
  int i = blockIdx.x * blockDim.x + threadIdx.x;
  if (i < BATCH * UNITS) { h32[i] = 0.0f; h16[i] = (_Float16)0.0f; }
}

// ---------------- pack W (K x 3072 row-major fp32) into WMMA-B-native f16 ----------------
// Record layout: for record r = nt*(K/32)+kc, lane L (0..31) holds 16 contiguous f16:
//   e<8  -> B[k0 + (L/16)*8 + e      ][nt*16 + (L%16)]
//   e>=8 -> B[k0 + 16 + (L/16)*8 + e-8][nt*16 + (L%16)]
// matching the 16-bit B operand VGPR layout of v_wmma_f32_16x16x32_f16.
__global__ void k_pack_b(const float* __restrict__ W, _Float16* __restrict__ Bp, int K) {
  const int N = GATES;
  int kChunks = K >> 5;
  int tid = blockIdx.x * blockDim.x + threadIdx.x;
  int total = (N / 16) * kChunks * 32;
  if (tid >= total) return;
  int lane = tid & 31;
  int rec  = tid >> 5;
  int kc = rec % kChunks;
  int nt = rec / kChunks;
  int half = lane >> 4;
  int n  = nt * 16 + (lane & 15);
  int k0 = kc * 32;
  _Float16* dst = Bp + (size_t)tid * 16;
#pragma unroll
  for (int e = 0; e < 16; ++e) {
    int koff = (e < 8) ? (half * 8 + e) : (16 + half * 8 + (e - 8));
    dst[e] = (_Float16)W[(size_t)(k0 + koff) * N + n];
  }
}

// ---------------- dual WMMA GEMM: C[128 x 3072] = A[128 x K](f16) @ Bpacked(f16) ----------------
// grid = (24, 8, 2); block = 256 (8 waves). Wave w of block x computes N-tile (x*8+w), M-tile y.
// z==0: rec = h @ Wh ; z==1: xg = x_t @ Wx
__global__ void k_gemm2(const _Float16* __restrict__ A0, long long lda0,
                        const _Float16* __restrict__ B0, float* __restrict__ C0, int K0,
                        const _Float16* __restrict__ A1, long long lda1,
                        const _Float16* __restrict__ B1, float* __restrict__ C1, int K1)
{
  const int N = GATES;
  const _Float16* A; const _Float16* Bp; float* C; long long lda; int K;
  if (blockIdx.z == 0) { A = A0; lda = lda0; Bp = B0; C = C0; K = K0; }
  else                 { A = A1; lda = lda1; Bp = B1; C = C1; K = K1; }

  int lane = threadIdx.x & 31;
  int wid  = threadIdx.x >> 5;
  int nt = blockIdx.x * 8 + wid;   // 0..191
  int mt = blockIdx.y;             // 0..7
  int half = lane >> 4;
  int m = mt * 16 + (lane & 15);
  int kChunks = K >> 5;

  f32x8 c = {};
  // A operand (16-bit, 16x32): lane holds row m; VGPR0-3 <- K = half*8 + 0..7,
  // VGPR4-7 <- K = 16 + half*8 + 0..7 : two contiguous 16B loads per chunk.
  const _Float16* arow = A + (long long)m * lda + half * 8;
  // Packed B: contiguous 32B per lane per chunk.
  const _Float16* bp = Bp + ((size_t)nt * kChunks * 32 + lane) * 16;

  for (int kc = 0; kc < kChunks; ++kc) {
    f16x8 alo = *(const f16x8*)(arow);
    f16x8 ahi = *(const f16x8*)(arow + 16);
    arow += 32;
    f16x16 a;
#pragma unroll
    for (int i = 0; i < 8; ++i) { a[i] = alo[i]; a[i + 8] = ahi[i]; }
    f16x16 b = *(const f16x16*)bp;
    bp += 512;  // 32 lanes * 16 f16
    c = __builtin_amdgcn_wmma_f32_16x16x32_f16(false, a, false, b, (short)0, c, false, false);
  }

  // D layout: VGPR v -> row mt*16 + v + 8*(lane/16), col nt*16 + lane%16
  float* crow = C + (size_t)(mt * 16 + 8 * half) * N + nt * 16 + (lane & 15);
#pragma unroll
  for (int v = 0; v < 8; ++v) crow[(size_t)v * N] = c[v];
}

// ---------------- fused GRU gates (Keras v2, reset_after=True) ----------------
__global__ void k_gates(const float* __restrict__ xg, const float* __restrict__ rec,
                        const float* __restrict__ bx, const float* __restrict__ bh,
                        float* __restrict__ h32, _Float16* __restrict__ h16,
                        _Float16* __restrict__ seq_out, int t)
{
  int idx = blockIdx.x * blockDim.x + threadIdx.x;  // b*1024 + u
  if (idx >= BATCH * UNITS) return;
  int b = idx >> 10;
  int u = idx & (UNITS - 1);
  size_t base = (size_t)b * GATES;

  float xz = xg[base + u]             + bx[u];
  float xr = xg[base + UNITS + u]     + bx[UNITS + u];
  float xh = xg[base + 2 * UNITS + u] + bx[2 * UNITS + u];
  float rz = rec[base + u]             + bh[u];
  float rr = rec[base + UNITS + u]     + bh[UNITS + u];
  float rh = rec[base + 2 * UNITS + u] + bh[2 * UNITS + u];

  float z  = 1.0f / (1.0f + __expf(-(xz + rz)));
  float r  = 1.0f / (1.0f + __expf(-(xr + rr)));
  float hh = tanhf(xh + r * rh);
  float hn = z * h32[idx] + (1.0f - z) * hh;

  h32[idx] = hn;
  h16[idx] = (_Float16)hn;
  if (seq_out)  // [B, T, U]
    seq_out[(size_t)b * TSTEPS * UNITS + (size_t)t * UNITS + u] = (_Float16)hn;
}

// ---------------- final dense head: out[b] = h[b,:] . Wd + bd ----------------
__global__ void k_dense(const float* __restrict__ h32, const float* __restrict__ Wd,
                        const float* __restrict__ bd, float* __restrict__ out)
{
  __shared__ float red[256];
  int b = blockIdx.x;
  float s = 0.0f;
  for (int u = threadIdx.x; u < UNITS; u += 256) s += h32[(size_t)b * UNITS + u] * Wd[u];
  red[threadIdx.x] = s;
  __syncthreads();
  for (int off = 128; off > 0; off >>= 1) {
    if (threadIdx.x < off) red[threadIdx.x] += red[threadIdx.x + off];
    __syncthreads();
  }
  if (threadIdx.x == 0) out[b] = red[0] + bd[0];
}

extern "C" void kernel_launch(void* const* d_in, const int* in_sizes, int n_in,
                              void* d_out, int out_size, void* d_ws, size_t ws_size,
                              hipStream_t stream) {
  (void)in_sizes; (void)n_in; (void)out_size; (void)ws_size;

  const float* x_in  = (const float*)d_in[0];
  const float* Wx[4] = {(const float*)d_in[1],  (const float*)d_in[5],
                        (const float*)d_in[9],  (const float*)d_in[13]};
  const float* Wh[4] = {(const float*)d_in[2],  (const float*)d_in[6],
                        (const float*)d_in[10], (const float*)d_in[14]};
  const float* bx[4] = {(const float*)d_in[3],  (const float*)d_in[7],
                        (const float*)d_in[11], (const float*)d_in[15]};
  const float* bh[4] = {(const float*)d_in[4],  (const float*)d_in[8],
                        (const float*)d_in[12], (const float*)d_in[16]};
  const float* Wd = (const float*)d_in[17];
  const float* bd = (const float*)d_in[18];
  float* out = (float*)d_out;

  // ---- workspace carve-out (~192 MB, all offsets 256B-aligned) ----
  char* p = (char*)d_ws;
  auto alloc = [&](size_t bytes) -> void* {
    void* r = (void*)p;
    p += (bytes + 255) & ~(size_t)255;
    return r;
  };
  _Float16* Whp[4];
  _Float16* Wxp[4];
  const int Kx[4] = {INDIM, UNITS, UNITS, UNITS};
  for (int l = 0; l < 4; ++l) Whp[l] = (_Float16*)alloc((size_t)UNITS * GATES * 2);
  for (int l = 0; l < 4; ++l) Wxp[l] = (_Float16*)alloc((size_t)Kx[l] * GATES * 2);
  _Float16* xseq = (_Float16*)alloc((size_t)BATCH * TSTEPS * INDIM * 2);
  _Float16* seqA = (_Float16*)alloc((size_t)BATCH * TSTEPS * UNITS * 2);
  _Float16* seqB = (_Float16*)alloc((size_t)BATCH * TSTEPS * UNITS * 2);
  float*    rec  = (float*)alloc((size_t)BATCH * GATES * 4);
  float*    xg   = (float*)alloc((size_t)BATCH * GATES * 4);
  float*    h32  = (float*)alloc((size_t)BATCH * UNITS * 4);
  _Float16* h16  = (_Float16*)alloc((size_t)BATCH * UNITS * 2);

  // ---- one-time conversions / packing ----
  {
    int n = BATCH * TSTEPS * INDIM;
    k_f32_to_f16<<<(n + 255) / 256, 256, 0, stream>>>(x_in, xseq, n);
  }
  for (int l = 0; l < 4; ++l) {
    int totH = (GATES / 16) * (UNITS >> 5) * 32;
    k_pack_b<<<(totH + 255) / 256, 256, 0, stream>>>(Wh[l], Whp[l], UNITS);
    int totX = (GATES / 16) * (Kx[l] >> 5) * 32;
    k_pack_b<<<(totX + 255) / 256, 256, 0, stream>>>(Wx[l], Wxp[l], Kx[l]);
  }

  // ---- 4 GRU layers ----
  const _Float16* seqIn[4] = {xseq, seqA, seqB, seqA};
  _Float16*       seqOut[4] = {seqA, seqB, seqA, nullptr};
  const long long ldaX[4] = {(long long)TSTEPS * INDIM, (long long)TSTEPS * UNITS,
                             (long long)TSTEPS * UNITS, (long long)TSTEPS * UNITS};

  dim3 ggrid(24, 8, 2);   // 192 N-tiles x 8 M-tiles, z: rec / xg
  for (int l = 0; l < 4; ++l) {
    k_zero_h<<<(BATCH * UNITS + 255) / 256, 256, 0, stream>>>(h32, h16);
    for (int t = 0; t < TSTEPS; ++t) {
      const _Float16* xbase = seqIn[l] + (size_t)t * Kx[l];
      k_gemm2<<<ggrid, 256, 0, stream>>>(h16, (long long)UNITS, Whp[l], rec, UNITS,
                                         xbase, ldaX[l],        Wxp[l], xg,  Kx[l]);
      k_gates<<<(BATCH * UNITS + 255) / 256, 256, 0, stream>>>(
          xg, rec, bx[l], bh[l], h32, h16, seqOut[l], t);
    }
  }

  // ---- dense head ----
  k_dense<<<BATCH, 256, 0, stream>>>(h32, Wd, bd, out);
}
